// F4_31568009626214
// MI455X (gfx1250) — matrix operand order repaired and checked by reading, not compiled, not run on hardware
//
#include <hip/hip_runtime.h>
#include <hip/hip_bf16.h>
#include <cstddef>

// ---------------------------------------------------------------------------
// Forest-of-trees routing + value accumulation for MI455X (gfx1250).
//
// Roofline: ~6.4 GFLOP vs >=384MB (realistically ~12GB of gathers, L2-heavy)
// => gather-bandwidth bound. Design:
//   * 1 block = 16-sample tile, 256 threads = 8 waves (wave32).
//   * x tile staged once in LDS as packed bf16 (32KB) -> WMMA A fragments.
//   * wave w == tree w: per level, gathers its 16 key rows straight from
//     global into B fragments (per-lane b128 loads + v_perm_b32 f32->bf16
//     pair packing), runs a K=1024 chain of v_wmma_f32_16x16x32_bf16,
//     extracts diag(X*K^T) per the ISA C/D layout, updates nodes in-lane,
//     prefetches the value row (global_prefetch_b8).
//   * all 8 waves then stream the gathered value rows (b128) into
//     register-resident y accumulators (64 f32 per thread).
// ---------------------------------------------------------------------------

#define BATCH     16384
#define D_IN      1024
#define D_OUT     1024
#define NUM_TREES 8
#define DEPTH     12
#define TILE      16          // samples per block
#define NBLOCKS   (BATCH / TILE)

typedef __attribute__((ext_vector_type(16))) __bf16 v16bf;
typedef __attribute__((ext_vector_type(8)))  float  v8f;

union AFrag { v16bf v; uint4 q[2]; unsigned int u[8]; };
union BFrag { v16bf v; uint4 q[2]; unsigned int u[8]; };

// pack two f32 into one dword of bf16 pair: low16 = bf16(lo), high16 = bf16(hi)
// (single v_perm_b32; truncation rounding — diag sign decisions unaffected in
//  the common case and the workload is gather-bound, not precision-bound)
static __device__ __forceinline__ unsigned int pk_bf16(float hi, float lo) {
  return __builtin_amdgcn_perm(__builtin_bit_cast(unsigned, hi),
                               __builtin_bit_cast(unsigned, lo),
                               0x07060302u);
}

__global__ void __launch_bounds__(256)
forest_fused_kernel(const float* __restrict__ x,
                    const float* __restrict__ keys,
                    const float* __restrict__ values,
                    float* __restrict__ y) {
  __shared__ unsigned int xs[TILE * D_IN / 2];  // 32 KB packed-bf16 x tile
  __shared__ float lam_s[NUM_TREES][TILE];
  __shared__ int   vnode_s[NUM_TREES][TILE];    // node used for value gather
  __shared__ int   nodes_s[NUM_TREES][TILE];    // current node per (tree,sample)

  const int tile = blockIdx.x;
  const int tid  = threadIdx.x;
  const int lane = tid & 31;
  const int w    = tid >> 5;                    // wave id == tree id (0..7)

  // ---- stage x tile: f32 -> packed bf16 -> LDS (64 floats per thread) ----
  {
    const float* xr = x + (size_t)tile * TILE * D_IN + tid * 64;
    const int ubase = tid * 32;
#pragma unroll
    for (int i = 0; i < 16; ++i) {
      float4 f = *reinterpret_cast<const float4*>(xr + 4 * i);
      xs[ubase + 2 * i + 0] = pk_bf16(f.y, f.x);
      xs[ubase + 2 * i + 1] = pk_bf16(f.w, f.z);
    }
  }
  if (tid < NUM_TREES * TILE) nodes_s[tid >> 4][tid & 15] = 0;
  __syncthreads();

  // ---- per-thread output accumulator: sample s, 64 columns at g*64 ----
  const int s    = tid >> 4;
  const int g    = tid & 15;
  const int col0 = g * 64;
  float acc[64];
#pragma unroll
  for (int i = 0; i < 64; ++i) acc[i] = 0.0f;

  // WMMA fragment lane geometry (16-bit A 16x32 / B 32x16, wave32)
  const int m     = lane & 15;                  // A: row M, B: column N (sample)
  const int abase = (lane < 16) ? 0 : 16;       // A K-offset within chunk
  const int kboff = (lane < 16) ? 0 : 16;       // B K-offset within chunk

  for (int level = 0; level < DEPTH; ++level) {
    // ================= routing: one wave per tree, full EXEC =================
    {
      const int node = nodes_s[w][m];
      const float* krow = keys + ((size_t)(node * NUM_TREES + w)) * D_IN;

      v8f c = {};
#pragma unroll 4
      for (int kc = 0; kc < D_IN; kc += 32) {
        AFrag a;   // x[m, kc+abase .. +15]
        a.q[0] = *reinterpret_cast<const uint4*>(&xs[(m * D_IN + kc + abase) >> 1]);
        a.q[1] = *reinterpret_cast<const uint4*>(&xs[(m * D_IN + kc + abase + 8) >> 1]);

        BFrag b;   // key[sample-m row][kc+kboff .. +15], packed to bf16 pairs
        const float4* kp = reinterpret_cast<const float4*>(krow + kc + kboff);
#pragma unroll
        for (int q = 0; q < 4; ++q) {
          float4 f = kp[q];
          b.u[2 * q + 0] = pk_bf16(f.y, f.x);
          b.u[2 * q + 1] = pk_bf16(f.w, f.z);
        }

        c = __builtin_amdgcn_wmma_f32_16x16x32_bf16(
                false, a.v, false, b.v, (short)0, c, false, false);
      }

      // diag(X*K^T): sample mm<8 -> lane mm, c[mm]; mm>=8 -> lane mm+16, c[mm-8]
      float cr[8];
#pragma unroll
      for (int i = 0; i < 8; ++i) cr[i] = c[i];
      const int idx = (lane < 8) ? lane : (lane - 24);
      float lam = cr[0];
#pragma unroll
      for (int i = 1; i < 8; ++i) lam = (idx == i) ? cr[i] : lam;

      const bool isdiag = (lane < 8) || (lane >= 24);
      if (isdiag) {
        const int mm  = (lane < 8) ? lane : (lane - 16);
        const int nod = nodes_s[w][mm];
        vnode_s[w][mm] = nod;
        lam_s[w][mm]   = lam;
        nodes_s[w][mm] = 2 * nod + 1 + (lam > 0.0f ? 1 : 0);
        // warm L2 for the value row we are about to stream
        __builtin_prefetch(values + ((size_t)(nod * NUM_TREES + w)) * D_OUT, 0, 0);
      }
    }
    __syncthreads();

    // ============ accumulation: all 256 threads stream value rows ============
#pragma unroll
    for (int k = 0; k < NUM_TREES; ++k) {
      const float lam = lam_s[k][s];
      const int   nod = vnode_s[k][s];
      const float4* vr = reinterpret_cast<const float4*>(
          values + ((size_t)(nod * NUM_TREES + k)) * D_OUT + col0);
#pragma unroll
      for (int q = 0; q < 16; ++q) {
        float4 v = vr[q];
        acc[4 * q + 0] = fmaf(lam, v.x, acc[4 * q + 0]);
        acc[4 * q + 1] = fmaf(lam, v.y, acc[4 * q + 1]);
        acc[4 * q + 2] = fmaf(lam, v.z, acc[4 * q + 2]);
        acc[4 * q + 3] = fmaf(lam, v.w, acc[4 * q + 3]);
      }
    }
    __syncthreads();  // lam_s/vnode_s consumed before next level overwrites
  }

  // ---- write y tile (each thread owns 64 contiguous columns of sample s) ----
  float* yo = y + ((size_t)(tile * TILE + s)) * D_OUT + col0;
#pragma unroll
  for (int q = 0; q < 16; ++q) {
    float4 o;
    o.x = acc[4 * q + 0];
    o.y = acc[4 * q + 1];
    o.z = acc[4 * q + 2];
    o.w = acc[4 * q + 3];
    *reinterpret_cast<float4*>(yo + 4 * q) = o;
  }
}

extern "C" void kernel_launch(void* const* d_in, const int* in_sizes, int n_in,
                              void* d_out, int out_size, void* d_ws, size_t ws_size,
                              hipStream_t stream) {
  (void)in_sizes; (void)n_in; (void)out_size; (void)d_ws; (void)ws_size;
  const float* x      = (const float*)d_in[0];
  const float* keys   = (const float*)d_in[1];
  const float* values = (const float*)d_in[2];
  float*       y      = (float*)d_out;

  forest_fused_kernel<<<dim3(NBLOCKS), dim3(256), 0, stream>>>(x, keys, values, y);
}
